// LSTMAttentionCell_2637109920064
// MI455X (gfx1250) — compile-verified, matrix operand order checked by reading, not run on hardware
//
#include <hip/hip_runtime.h>
#include <hip/hip_bf16.h>
#include <math.h>

typedef __bf16 bf16;
typedef __attribute__((ext_vector_type(16))) __bf16 v16bf;
typedef __attribute__((ext_vector_type(8)))  __bf16 v8bf;
typedef __attribute__((ext_vector_type(8)))  float  v8f;

// Problem constants
constexpr int Bc   = 64;    // batch
constexpr int Tc   = 512;   // time steps
constexpr int Dc   = 3;     // x feature dim
constexpr int Hc   = 512;   // hidden
constexpr int G4H  = 2048;  // 4*H gate width
constexpr int Kmix = 10;    // mixture components
constexpr int Lc   = 64;    // attention length
constexpr int Wc   = 80;    // attention value width
// Packed K dims (multiples of 32)
constexpr int K1c  = 608;   // actA: [w(80), x(3), pad13, h1(512)]
constexpr int K2c  = 1120;  // act2/3: [x(3), h(512), w(80), pad13, hrec(512)]
constexpr int NAc  = 32;    // attention N padded (30 -> 32)
constexpr int NWG  = 64;    // persistent workgroups (== Bc)
constexpr int NTHR = 256;   // 8 waves of 32

constexpr unsigned SMEM_BYTES = (unsigned)Bc * K2c * sizeof(bf16);  // 143,360 B

extern __shared__ __align__(16) bf16 smem[];   // activation panel staging

// ---------------------------------------------------------------------------
// WMMA fragment load: per ISA 16-bit A/B layout, lane (kh=lane/16) needs
// k in [8kh, 8kh+8) and [16+8kh, 24+8kh)  -> two contiguous 16B loads.
__device__ inline v16bf ld_frag(const bf16* p) {
    union { v16bf v; v8bf h[2]; } u;
    u.h[0] = *(const v8bf*)(p);       // k = 8kh .. 8kh+7
    u.h[1] = *(const v8bf*)(p + 16);  // k = 16+8kh .. 23+8kh
    return u.v;
}

// Cooperative async stage of an activation panel (global bf16 -> LDS).
// elems must be a multiple of 8*NTHR (64*608 and 64*1120 both are).
// Uses GLOBAL_LOAD_ASYNC_TO_LDS_B128 (ASYNCcnt) per the CDNA5 ISA; the
// low 32 bits of a generic LDS address are the workgroup-relative offset.
__device__ inline void stage_act(const bf16* __restrict__ act, int elems) {
    __syncthreads();                       // smem no longer read by prev phase
    const int nvec = elems >> 3;           // 16-byte chunks
    for (int i = threadIdx.x; i < nvec; i += NTHR) {
        const unsigned loff = (unsigned)(size_t)(smem + (size_t)i * 8);
        const bf16* g = act + (size_t)i * 8;
        asm volatile("global_load_async_to_lds_b128 %0, %1, off"
                     :: "v"(loff), "v"(g) : "memory");
    }
    asm volatile("s_wait_asynccnt 0x0" ::: "memory");
    __syncthreads();
}

// One 64 x 32 output slice of  Z = actLDS[64,K] * panel[K,Ntot] (bf16 -> f32).
// A fragments come from the LDS-staged panel (ds_load_b128 pairs); B panel is
// streamed from global (L2-resident) with element (k,n) at
// ((k/32)*Ntot + n)*32 + k%32.  8 waves: wave = (mt*2 + nt), one 16x16 tile.
__device__ inline void gemm_slice(const bf16* __restrict__ actL, int actStride,
                                  const bf16* __restrict__ panel, int Ntot,
                                  float* __restrict__ z, int zStride,
                                  int kchunks, int colbase) {
    const int wave = threadIdx.x >> 5;
    const int lane = threadIdx.x & 31;
    const int mt = wave >> 1;       // 0..3
    const int nt = wave & 1;        // 0..1
    const int lr = lane & 15;
    const int kh = lane >> 4;

    v8f acc = {0.f, 0.f, 0.f, 0.f, 0.f, 0.f, 0.f, 0.f};
    const bf16* aP = actL + (size_t)(mt * 16 + lr) * actStride + 8 * kh;
    const bf16* bP = panel + (size_t)(colbase + nt * 16 + lr) * 32 + 8 * kh;
    const size_t bStep = (size_t)Ntot * 32;

    for (int kc = 0; kc < kchunks; ++kc) {
        __builtin_prefetch(bP + 2 * bStep, 0, 3);   // global_prefetch_b8 (near)
        v16bf a = ld_frag(aP);                      // ds_load_b128 x2
        v16bf b = ld_frag(bP);                      // global_load_b128 x2
        acc = __builtin_amdgcn_wmma_f32_16x16x32_bf16(
            false, a, false, b, (short)0, acc, false, false);
        aP += 32;
        bP += bStep;
    }
    // C/D layout: VGPR r on lane l holds (M = r + 8*(l>>4), N = l&15)
    const int m0 = mt * 16 + kh * 8;
    const int n  = colbase + nt * 16 + lr;
#pragma unroll
    for (int r = 0; r < 8; ++r)
        z[(size_t)(m0 + r) * zStride + n] = acc[r];
}

// ---------------------------------------------------------------------------
__device__ inline float sigmoidf_(float x) { return 1.0f / (1.0f + __expf(-x)); }
__device__ inline float softplusf_(float x) {
    return (x > 20.f) ? x : log1pf(__expf(x));
}

// Global sense barrier (monotonic counter).
__device__ inline void gbar(unsigned* cnt, unsigned& phase) {
    __syncthreads();
    __threadfence();
    ++phase;
    if (threadIdx.x == 0) {
        atomicAdd(cnt, 1u);
        const unsigned target = phase * NWG;
        while (*(volatile unsigned*)cnt < target) __builtin_amdgcn_s_sleep(1);
    }
    __syncthreads();
    __threadfence();
}

// Gate pointwise: z[64,2048] (+bias) -> c,h ; h written as bf16 into two
// packed activation buffers, optionally f32 into output.
__device__ inline void lstm_pointwise(const float* __restrict__ z,
                                      const float* __restrict__ bias,
                                      float* __restrict__ c,
                                      bf16* hA, int strideA, int offA,
                                      bf16* hB, int strideB, int offB,
                                      float* hout) {
    const int gtid = blockIdx.x * NTHR + threadIdx.x;
    for (int e = gtid; e < Bc * Hc; e += NWG * NTHR) {
        const int b = e >> 9;       // /512
        const int j = e & 511;
        const size_t r = (size_t)b * G4H;
        const float zi = z[r + j]          + bias[j];
        const float zf = z[r + 512 + j]    + bias[512 + j];
        const float zg = z[r + 1024 + j]   + bias[1024 + j];
        const float zo = z[r + 1536 + j]   + bias[1536 + j];
        const float ig = sigmoidf_(zi);
        const float fg = sigmoidf_(zf);
        const float gg = tanhf(zg);
        const float og = sigmoidf_(zo);
        const float cn = fg * c[e] + ig * gg;
        const float h  = og * tanhf(cn);
        c[e] = cn;
        const bf16 hb = (bf16)h;
        hA[(size_t)b * strideA + offA + j] = hb;
        hB[(size_t)b * strideB + offB + j] = hb;
        if (hout) hout[(size_t)b * ((size_t)Tc * Hc) + j] = h;
    }
}

// Attention (workgroup 0 only): re-stage actA (fresh h1), ap = actA @ Pa,
// softplus, kappa update, phi.
__device__ inline void attention_phase(const bf16* actA, const bf16* panelA,
                                       const float* ba, float* ap,
                                       float* alpha, float* beta,
                                       float* kappa, float* phi) {
    stage_act(actA, Bc * K1c);
    gemm_slice(smem, K1c, panelA, NAc, ap, NAc, K1c / 32, 0);
    __syncthreads();
    for (int e = threadIdx.x; e < Bc * Kmix; e += NTHR) {
        const int b = e / Kmix, k = e % Kmix;
        const float* r = ap + (size_t)b * NAc;
        const float a  = softplusf_(r[k]          + ba[k]);
        const float be = softplusf_(r[10 + k]     + ba[10 + k]);
        const float dk = softplusf_(r[20 + k]     + ba[20 + k]);
        const float kp = kappa[e] + dk * 0.04f;   // /25
        kappa[e] = kp;
        alpha[e] = a;
        beta[e]  = fmaxf(be, 0.01f);
    }
    __syncthreads();
    for (int e = threadIdx.x; e < Bc * Lc; e += NTHR) {
        const int b = e / Lc, l = e % Lc;
        const float u = (float)l;
        float s = 0.f;
#pragma unroll
        for (int k = 0; k < Kmix; ++k) {
            const float d = kappa[b * Kmix + k] - u;
            s += alpha[b * Kmix + k] * __expf(-(d * d) / beta[b * Kmix + k]);
        }
        phi[e] = s;
    }
}

// w[b,:] = phi[b,:] @ masked_vals[b,:,:]  ; one workgroup per batch row.
__device__ inline void w_phase(const float* __restrict__ phi,
                               const float* __restrict__ mv,
                               bf16* actA, bf16* act2, bf16* act3) {
    const int b = blockIdx.x;
    __shared__ float sphi[Lc];
    if (threadIdx.x < Lc) sphi[threadIdx.x] = phi[b * Lc + threadIdx.x];
    __syncthreads();
    for (int wcol = threadIdx.x; wcol < Wc; wcol += NTHR) {
        float s = 0.f;
        const float* m = mv + (size_t)b * Lc * Wc + wcol;
#pragma unroll 4
        for (int l = 0; l < Lc; ++l) s += sphi[l] * m[(size_t)l * Wc];
        const bf16 wb = (bf16)s;
        actA[(size_t)b * K1c + wcol]       = wb;   // actA: w at col 0
        act2[(size_t)b * K2c + 515 + wcol] = wb;   // act2: [x,h1,w...]
        act3[(size_t)b * K2c + 515 + wcol] = wb;
    }
    __syncthreads();
}

// ---------------------------------------------------------------------------
// Weight packing into WMMA B-panel layout (runs once per launch).
// packed row k: k<inLen -> Wk[k] ; inPad<=k<inPad+512 -> Wr[k-inPad] ; else 0.
__global__ void pack_panel(const float* __restrict__ Wk,
                           const float* __restrict__ Wr,
                           bf16* __restrict__ panel,
                           int inLen, int inPad, int Ktot,
                           int Nsrc, int Ntot, int ldsrc) {
    const size_t total = (size_t)Ktot * Ntot;
    const size_t gs = (size_t)gridDim.x * blockDim.x;
    for (size_t i = (size_t)blockIdx.x * blockDim.x + threadIdx.x; i < total; i += gs) {
        const int k = (int)(i / Ntot);
        const int n = (int)(i % Ntot);
        float v = 0.f;
        if (n < Nsrc) {
            if (k < inLen)                         v = Wk[(size_t)k * ldsrc + n];
            else if (k >= inPad && k < inPad + Hc) v = Wr[(size_t)(k - inPad) * ldsrc + n];
        }
        panel[(((size_t)(k >> 5)) * Ntot + n) * 32 + (k & 31)] = (bf16)v;
    }
}

__global__ void init_kernel(const float* __restrict__ attn,
                            const int* __restrict__ lengths,
                            bf16* actA, bf16* act2, bf16* act3,
                            float* c1, float* c2, float* c3,
                            float* kappa, float* mv, unsigned* cnt) {
    const size_t gt = (size_t)blockIdx.x * blockDim.x + threadIdx.x;
    const size_t gs = (size_t)gridDim.x * blockDim.x;
    for (size_t i = gt; i < (size_t)Bc * K1c; i += gs) actA[i] = (bf16)0.f;
    for (size_t i = gt; i < (size_t)Bc * K2c; i += gs) { act2[i] = (bf16)0.f; act3[i] = (bf16)0.f; }
    for (size_t i = gt; i < (size_t)Bc * Hc; i += gs) { c1[i] = 0.f; c2[i] = 0.f; c3[i] = 0.f; }
    for (size_t i = gt; i < (size_t)Bc * Kmix; i += gs) kappa[i] = 0.f;
    for (size_t i = gt; i < (size_t)Bc * Lc * Wc; i += gs) {
        const size_t b = i / ((size_t)Lc * Wc);
        const size_t l = (i / Wc) % Lc;
        mv[i] = attn[i] * ((l < (size_t)lengths[b]) ? 1.f : 0.f);
    }
    if (gt == 0) *cnt = 0u;
}

// Seed x_0 into the packed activation buffers (after zeroing).
__global__ void seed_kernel(const float* __restrict__ x,
                            bf16* actA, bf16* act2, bf16* act3) {
    if (threadIdx.x < Bc * Dc) {
        const int b = threadIdx.x / Dc, d = threadIdx.x % Dc;
        const bf16 xv = (bf16)x[(size_t)b * Tc * Dc + d];   // t = 0
        actA[(size_t)b * K1c + 80 + d] = xv;
        act2[(size_t)b * K2c + d]      = xv;
        act3[(size_t)b * K2c + d]      = xv;
    }
}

// ---------------------------------------------------------------------------
// Persistent kernel: full T-step recurrence with global barriers.
__global__ __launch_bounds__(NTHR)
void graves_kernel(const float* __restrict__ x,
                   bf16* actA, bf16* act2, bf16* act3,
                   const bf16* P1, const bf16* P2, const bf16* P3, const bf16* Pa,
                   float* z, float* ap,
                   const float* b1, const float* b2, const float* b3, const float* ba,
                   float* c1, float* c2, float* c3,
                   float* kappa, float* alpha, float* beta, float* phi,
                   const float* mv, float* out, unsigned* cnt) {
    unsigned phase = 0;
    const int colbase = blockIdx.x * 32;

    for (int t = 0; t < Tc; ++t) {
        // LSTM1:  z = [w,x,h1] @ P1   (stage actA -> LDS, WMMA from LDS)
        stage_act(actA, Bc * K1c);
        gemm_slice(smem, K1c, P1, G4H, z, G4H, K1c / 32, colbase);
        gbar(cnt, phase);
        lstm_pointwise(z, b1, c1, actA, K1c, 96, act2, K2c, 3, nullptr);
        gbar(cnt, phase);

        // Attention mixture (uses fresh h1, old w)
        if (blockIdx.x == 0)
            attention_phase(actA, Pa, ba, ap, alpha, beta, kappa, phi);
        gbar(cnt, phase);
        w_phase(phi, mv, actA, act2, act3);
        gbar(cnt, phase);

        // LSTM2:  z = [x,h1,w,h2] @ P2
        stage_act(act2, Bc * K2c);
        gemm_slice(smem, K2c, P2, G4H, z, G4H, K2c / 32, colbase);
        gbar(cnt, phase);
        lstm_pointwise(z, b2, c2, act2, K2c, 608, act3, K2c, 3, nullptr);
        gbar(cnt, phase);

        // LSTM3:  z = [x,h2,w,h3] @ P3
        stage_act(act3, Bc * K2c);
        gemm_slice(smem, K2c, P3, G4H, z, G4H, K2c / 32, colbase);
        gbar(cnt, phase);
        lstm_pointwise(z, b3, c3, act3, K2c, 608, act3, K2c, 608,
                       out + (size_t)t * Hc);

        // Stage x_{t+1} for next step
        if (blockIdx.x == 0 && threadIdx.x < Bc * Dc && t + 1 < Tc) {
            const int b = threadIdx.x / Dc, d = threadIdx.x % Dc;
            const bf16 xv = (bf16)x[((size_t)b * Tc + (t + 1)) * Dc + d];
            actA[(size_t)b * K1c + 80 + d] = xv;
            act2[(size_t)b * K2c + d]      = xv;
            act3[(size_t)b * K2c + d]      = xv;
        }
        gbar(cnt, phase);
    }
}

// ---------------------------------------------------------------------------
extern "C" void kernel_launch(void* const* d_in, const int* in_sizes, int n_in,
                              void* d_out, int out_size, void* d_ws, size_t ws_size,
                              hipStream_t stream) {
    (void)in_sizes; (void)n_in; (void)out_size; (void)ws_size;
    const float* x   = (const float*)d_in[0];
    const float* av  = (const float*)d_in[1];
    const int*   len = (const int*)d_in[2];
    const float* Wk1 = (const float*)d_in[3];
    const float* Wr1 = (const float*)d_in[4];
    const float* b1  = (const float*)d_in[5];
    const float* Wa  = (const float*)d_in[6];
    const float* ba  = (const float*)d_in[7];
    const float* Wk2 = (const float*)d_in[8];
    const float* Wr2 = (const float*)d_in[9];
    const float* b2  = (const float*)d_in[10];
    const float* Wk3 = (const float*)d_in[11];
    const float* Wr3 = (const float*)d_in[12];
    const float* b3  = (const float*)d_in[13];
    float* out = (float*)d_out;

    char* p = (char*)d_ws;
    auto alloc = [&](size_t bytes) -> char* {
        char* r = p;
        p += (bytes + 255) & ~(size_t)255;
        return r;
    };
    bf16* P1   = (bf16*)alloc((size_t)K1c * G4H * sizeof(bf16));
    bf16* P2   = (bf16*)alloc((size_t)K2c * G4H * sizeof(bf16));
    bf16* P3   = (bf16*)alloc((size_t)K2c * G4H * sizeof(bf16));
    bf16* Pa   = (bf16*)alloc((size_t)K1c * NAc * sizeof(bf16));
    bf16* actA = (bf16*)alloc((size_t)Bc * K1c * sizeof(bf16));
    bf16* act2 = (bf16*)alloc((size_t)Bc * K2c * sizeof(bf16));
    bf16* act3 = (bf16*)alloc((size_t)Bc * K2c * sizeof(bf16));
    float* z     = (float*)alloc((size_t)Bc * G4H * sizeof(float));
    float* ap    = (float*)alloc((size_t)Bc * NAc * sizeof(float));
    float* c1    = (float*)alloc((size_t)Bc * Hc * sizeof(float));
    float* c2    = (float*)alloc((size_t)Bc * Hc * sizeof(float));
    float* c3    = (float*)alloc((size_t)Bc * Hc * sizeof(float));
    float* kappa = (float*)alloc((size_t)Bc * Kmix * sizeof(float));
    float* alpha = (float*)alloc((size_t)Bc * Kmix * sizeof(float));
    float* beta  = (float*)alloc((size_t)Bc * Kmix * sizeof(float));
    float* phi   = (float*)alloc((size_t)Bc * Lc * sizeof(float));
    float* mv    = (float*)alloc((size_t)Bc * Lc * Wc * sizeof(float));
    unsigned* cnt = (unsigned*)alloc(256);

    // Pack weights (bf16, WMMA B-panel layout). Attention's recurrent block
    // is just rows 83.. of Wa.
    pack_panel<<<dim3(1024), dim3(256), 0, stream>>>(Wk1, Wr1, P1, 83, 96, K1c, G4H, G4H, G4H);
    pack_panel<<<dim3(2048), dim3(256), 0, stream>>>(Wk2, Wr2, P2, 595, 608, K2c, G4H, G4H, G4H);
    pack_panel<<<dim3(2048), dim3(256), 0, stream>>>(Wk3, Wr3, P3, 595, 608, K2c, G4H, G4H, G4H);
    pack_panel<<<dim3(64),   dim3(256), 0, stream>>>(Wa, Wa + (size_t)83 * 30, Pa, 83, 96, K1c, 30, NAc, 30);

    init_kernel<<<dim3(512), dim3(256), 0, stream>>>(av, len, actA, act2, act3,
                                                     c1, c2, c3, kappa, mv, cnt);
    seed_kernel<<<dim3(1), dim3(256), 0, stream>>>(x, actA, act2, act3);

    graves_kernel<<<dim3(NWG), dim3(NTHR), SMEM_BYTES, stream>>>(
        x, actA, act2, act3, P1, P2, P3, Pa, z, ap,
        b1, b2, b3, ba, c1, c2, c3, kappa, alpha, beta, phi, mv, out, cnt);
}
